// BronxLayer_48722109006226
// MI455X (gfx1250) — compile-verified, matrix operand order checked by reading, not run on hardware
//
#include <hip/hip_runtime.h>
#include <hip/hip_bf16.h>
#include <math.h>

#define N 8192
#define D 256
#define ND (N * D)
#define CHUNKS 16
#define JS 2

typedef __attribute__((ext_vector_type(16))) __bf16 v16bf;
typedef __attribute__((ext_vector_type(8)))  float  v8f;

typedef unsigned short u16;
typedef unsigned int   u32;

struct B32x2 { uint4 lo, hi; };

// float -> bf16 round-to-nearest-even (bit pattern)
__device__ __forceinline__ u16 f2bf(float f) {
    u32 u = __builtin_bit_cast(u32, f);
    u32 r = u + 0x7fffu + ((u >> 16) & 1u);
    return (u16)(r >> 16);
}

// Load one v16bf WMMA operand: lane's 16 bf16 values live at [p..p+7] and [p+16..p+23]
// (caller passes p = base + half*8; both chunks are 16B-aligned)
__device__ __forceinline__ v16bf ldbf16x16(const u16* p) {
    B32x2 t;
    t.lo = *(const uint4*)(p);
    t.hi = *(const uint4*)(p + 16);
    return __builtin_bit_cast(v16bf, t);
}

__device__ __forceinline__ v8f wmma_bf16(v16bf a, v16bf b, v8f c) {
    return __builtin_amdgcn_wmma_f32_16x16x32_bf16(
        false, a, false, b, (short)0, c, false, false);
}

// reductions across the 16 lanes of a wave32 half (xor masks stay within half)
__device__ __forceinline__ float hmax16(float v) {
    v = fmaxf(v, __shfl_xor(v, 1, 32));
    v = fmaxf(v, __shfl_xor(v, 2, 32));
    v = fmaxf(v, __shfl_xor(v, 4, 32));
    v = fmaxf(v, __shfl_xor(v, 8, 32));
    return v;
}
__device__ __forceinline__ float hsum16(float v) {
    v += __shfl_xor(v, 1, 32);
    v += __shfl_xor(v, 2, 32);
    v += __shfl_xor(v, 4, 32);
    v += __shfl_xor(v, 8, 32);
    return v;
}

// ---------------- LayerNorm + bf16 conversion of x ----------------
__global__ void k_prep(const float* __restrict__ h, const float* __restrict__ x,
                       const float* __restrict__ gamma, const float* __restrict__ beta,
                       u16* __restrict__ hn_b, u16* __restrict__ x_b) {
    int row = blockIdx.x;
    int t = threadIdx.x;
    size_t base = (size_t)row * D;
    float v = h[base + t];
    __shared__ float s1[256], s2[256];
    s1[t] = v; s2[t] = v * v;
    __syncthreads();
    for (int off = 128; off > 0; off >>= 1) {
        if (t < off) { s1[t] += s1[t + off]; s2[t] += s2[t + off]; }
        __syncthreads();
    }
    float mu  = s1[0] * (1.0f / D);
    float var = s2[0] * (1.0f / D) - mu * mu;
    float rs  = rsqrtf(var + 1e-5f);
    float hn  = (v - mu) * rs * gamma[t] + beta[t];
    hn_b[base + t] = f2bf(hn);
    x_b[base + t]  = f2bf(x[base + t]);
}

// ---------------- weight f32 -> bf16, TRANSPOSED: wt[n][k] = w[k][n] ----------------
__global__ void k_convwt(const float* __restrict__ w, u16* __restrict__ wt) {
    int i = blockIdx.x * blockDim.x + threadIdx.x;   // i = k*D + n
    int k = i >> 8, n = i & 255;
    wt[(size_t)n * D + k] = f2bf(w[i]);
}

// ---------------- 2x2 mixing softmax over dim 0 ----------------
__global__ void k_mixprep(const float* __restrict__ mixing, float* __restrict__ m) {
    if (threadIdx.x == 0) {
        for (int c = 0; c < 2; ++c) {
            float a = mixing[c], b = mixing[2 + c];
            float mx = fmaxf(a, b);
            float ea = __expf(a - mx), eb = __expf(b - mx);
            float s = ea + eb;
            m[c]     = ea / s;   // m[0][c]
            m[2 + c] = eb / s;   // m[1][c]
        }
    }
}

// ---------------- GEMM: out[M,256] = A_bf16[M,256] @ W (Wt transposed bf16) ----------
// mode 0: store bf16; mode 1: ELU(out) + res -> f32
__global__ void k_gemm256(const u16* __restrict__ A, const u16* __restrict__ Wt,
                          u16* __restrict__ outb, float* __restrict__ outf,
                          const float* __restrict__ res, int mode) {
    __shared__ __attribute__((aligned(16))) u16 sA[16 * 256];
    int t = threadIdx.x;
    int wave = t >> 5, lane = t & 31;
    int half = lane >> 4, col = lane & 15;
    int i0 = blockIdx.x * 16;
    { // stage 16 A rows cooperatively
        int r = t >> 4, seg = t & 15;
        const uint4* src = (const uint4*)(A + (size_t)(i0 + r) * D + seg * 16);
        uint4* dst = (uint4*)(sA + r * 256 + seg * 16);
        dst[0] = src[0]; dst[1] = src[1];
    }
    __syncthreads();
    v16bf Aa[8];
#pragma unroll
    for (int s = 0; s < 8; ++s)
        Aa[s] = ldbf16x16(sA + col * 256 + s * 32 + half * 8);
#pragma unroll
    for (int tt = 0; tt < 2; ++tt) {
        int n0 = (wave + tt * 8) * 16;
        v8f C = {};
#pragma unroll
        for (int s = 0; s < 8; ++s) {
            v16bf B = ldbf16x16(Wt + (size_t)(n0 + col) * D + s * 32 + half * 8);
            C = wmma_bf16(Aa[s], B, C);
        }
#pragma unroll
        for (int r = 0; r < 8; ++r) {
            int orow = r + half * 8;
            size_t idx = (size_t)(i0 + orow) * D + n0 + col;
            float v = C[r];
            if (mode == 0) {
                outb[idx] = f2bf(v);
            } else {
                v = v > 0.f ? v : expm1f(v);
                outf[idx] = v + res[idx];
            }
        }
    }
}

// ---------------- Pass 1: per-row softmax stats + gram row sum-of-squares ----------------
__global__ void k_pass1(const u16* __restrict__ kb, const u16* __restrict__ qb,
                        const u16* __restrict__ xb, float* __restrict__ statsp) {
    __shared__ __attribute__((aligned(16))) u16 sq[16 * 256];
    __shared__ __attribute__((aligned(16))) u16 sx[16 * 256];
    int t = threadIdx.x, wave = t >> 5, lane = t & 31;
    int half = lane >> 4, col = lane & 15;
    int itile = blockIdx.x * 8 + wave;
    int i0 = itile * 16;
    int chunk = blockIdx.y;

    v16bf Ak[8], Ax[8];
#pragma unroll
    for (int s = 0; s < 8; ++s) {
        Ak[s] = ldbf16x16(kb + (size_t)(i0 + col) * D + s * 32 + half * 8);
        Ax[s] = ldbf16x16(xb + (size_t)(i0 + col) * D + s * 32 + half * 8);
    }
    float rmax[8], rsum[8], gsq[8];
#pragma unroll
    for (int r = 0; r < 8; ++r) { rmax[r] = -3.0e38f; rsum[r] = 0.f; gsq[r] = 0.f; }
    const float scale = 0.0625f;  // D^-0.5 = 1/16
    int j0 = chunk * (N / CHUNKS);
    int j1 = j0 + (N / CHUNKS);
    for (int j = j0; j < j1; j += 16) {
        __syncthreads();
        { // stage q,x rows j..j+15
            int r = t >> 4, seg = t & 15;
            const uint4* s0 = (const uint4*)(qb + (size_t)(j + r) * D + seg * 16);
            uint4* d0 = (uint4*)(sq + r * 256 + seg * 16);
            d0[0] = s0[0]; d0[1] = s0[1];
            const uint4* s1 = (const uint4*)(xb + (size_t)(j + r) * D + seg * 16);
            uint4* d1 = (uint4*)(sx + r * 256 + seg * 16);
            d1[0] = s1[0]; d1[1] = s1[1];
        }
        __syncthreads();
        v8f S = {}, G = {};
#pragma unroll
        for (int s = 0; s < 8; ++s) {
            v16bf Bq = ldbf16x16(sq + col * 256 + s * 32 + half * 8);
            v16bf Bx = ldbf16x16(sx + col * 256 + s * 32 + half * 8);
            S = wmma_bf16(Ak[s], Bq, S);
            G = wmma_bf16(Ax[s], Bx, G);
        }
#pragma unroll
        for (int r = 0; r < 8; ++r) {
            float sv = S[r] * scale;
            float tm = hmax16(sv);
            float nm = fmaxf(rmax[r], tm);
            float e  = __expf(sv - nm);
            float es = hsum16(e);
            rsum[r] = rsum[r] * __expf(rmax[r] - nm) + es;
            rmax[r] = nm;
            float g = G[r];
            gsq[r] += hsum16(g * g);
        }
    }
    if (col == 0) {
#pragma unroll
        for (int r = 0; r < 8; ++r) {
            int row = i0 + r + half * 8;
            size_t o = ((size_t)row * CHUNKS + chunk) * 3;
            statsp[o + 0] = rmax[r];
            statsp[o + 1] = rsum[r];
            statsp[o + 2] = gsq[r];
        }
    }
}

// ---------------- merge per-chunk stats ----------------
__global__ void k_mstats(const float* __restrict__ statsp, float* __restrict__ statsf) {
    int row = blockIdx.x * blockDim.x + threadIdx.x;
    if (row >= N) return;
    float M = -3.0e38f;
    for (int c = 0; c < CHUNKS; ++c)
        M = fmaxf(M, statsp[((size_t)row * CHUNKS + c) * 3]);
    float L = 0.f, Gs = 0.f;
    for (int c = 0; c < CHUNKS; ++c) {
        size_t o = ((size_t)row * CHUNKS + c) * 3;
        L  += statsp[o + 1] * __expf(statsp[o + 0] - M);
        Gs += statsp[o + 2];
    }
    statsf[row * 3 + 0] = M;
    statsf[row * 3 + 1] = 1.0f / L;
    statsf[row * 3 + 2] = 1.0f / fmaxf(sqrtf(Gs), 1e-12f);
}

// ---------------- Pass 2: wave-pair fused apply ----------------
// Wave pair p = wave>>1 owns one 16-row i-tile. role 0: S=k.q^T path, accumulates
// wH @ hn into partH. role 1: G=x.x^T path, accumulates wX @ x into partX.
// S and G are exchanged through LDS; hn/x are staged transposed for the second GEMM.
__global__ void k_pass2(const u16* __restrict__ kb, const u16* __restrict__ qb,
                        const u16* __restrict__ xb, const u16* __restrict__ hnb,
                        const float* __restrict__ statsf, const float* __restrict__ mix,
                        float* __restrict__ partH, float* __restrict__ partX) {
    __shared__ __attribute__((aligned(16))) u16 sqj[32 * 256];   // q rows, row-major
    __shared__ __attribute__((aligned(16))) u16 sxj[32 * 256];   // x rows, row-major
    __shared__ __attribute__((aligned(16))) u16 shnT[256 * 32];  // hn cols x 32 rows
    __shared__ __attribute__((aligned(16))) u16 sxjT[256 * 32];  // x  cols x 32 rows
    __shared__ float sS[4 * 512];                                // raw S per pair
    __shared__ float sG[4 * 512];                                // raw G per pair
    __shared__ __attribute__((aligned(16))) u16 swW[8 * 512];    // weight A tiles
    int t = threadIdx.x, wave = t >> 5, lane = t & 31;
    int half = lane >> 4, col = lane & 15;
    int p = wave >> 1, role = wave & 1;
    int itile = blockIdx.x * 4 + p;
    int i0 = itile * 16;
    int js = blockIdx.y;

    const u16* Abase = role ? xb : kb;        // score A operand rows
    const u16* sb    = role ? sxj : sqj;      // score B operand rows
    const u16* sbt   = role ? sxjT : shnT;    // accumulation B (transposed)
    float* sdst = (role ? sG : sS) + p * 512; // my score tile slot
    float cP = role ? mix[1] : mix[0];        // m[0][1] : m[0][0]
    float cG = role ? mix[3] : mix[2];        // m[1][1] : m[1][0]
    float* pout = role ? partX : partH;
    u16* myW = swW + wave * 512;

    float sm[8], sli[8], sgi[8];
#pragma unroll
    for (int r = 0; r < 8; ++r) {
        int row = i0 + r + half * 8;
        sm[r]  = statsf[row * 3 + 0];
        sli[r] = statsf[row * 3 + 1];
        sgi[r] = statsf[row * 3 + 2];
    }
    v8f zacc = {};
    v8f acc[16];
#pragma unroll
    for (int tt = 0; tt < 16; ++tt) acc[tt] = zacc;
    const float scale = 0.0625f;
    int j0 = js * (N / JS), j1 = j0 + (N / JS);
    for (int j = j0; j < j1; j += 32) {
        __syncthreads();
        { // stage: q,x row-major; hn,x transposed
            int r = t >> 3, seg = t & 7;
            const uint4* sq_ = (const uint4*)(qb  + (size_t)(j + r) * D + seg * 32);
            const uint4* sx_ = (const uint4*)(xb  + (size_t)(j + r) * D + seg * 32);
            const uint4* sh_ = (const uint4*)(hnb + (size_t)(j + r) * D + seg * 32);
            uint4* dq = (uint4*)(sqj + r * 256 + seg * 32);
            uint4* dx = (uint4*)(sxj + r * 256 + seg * 32);
            uint4 vh[4], vx[4];
#pragma unroll
            for (int m = 0; m < 4; ++m) {
                uint4 q4 = sq_[m]; vh[m] = sh_[m]; vx[m] = sx_[m];
                dq[m] = q4; dx[m] = vx[m];
            }
            const u16* ph = (const u16*)vh;
            const u16* px = (const u16*)vx;
#pragma unroll
            for (int m = 0; m < 32; ++m) {
                shnT[(seg * 32 + m) * 32 + r] = ph[m];
                sxjT[(seg * 32 + m) * 32 + r] = px[m];
            }
        }
        __syncthreads();
        // score phase: role 0 -> S = k_i . q_j^T ; role 1 -> G = x_i . x_j^T
#pragma unroll
        for (int sub = 0; sub < 2; ++sub) {
            v8f S = {};
#pragma unroll
            for (int s = 0; s < 8; ++s) {
                v16bf Aa = ldbf16x16(Abase + (size_t)(i0 + col) * D + s * 32 + half * 8);
                v16bf B  = ldbf16x16(sb + (sub * 16 + col) * 256 + s * 32 + half * 8);
                S = wmma_bf16(Aa, B, S);
            }
#pragma unroll
            for (int r = 0; r < 8; ++r)
                sdst[sub * 256 + (r + half * 8) * 16 + col] = S[r];
        }
        __syncthreads();
        // weight phase: combine softmax(S) and normalized G, stage bf16 A tile
#pragma unroll
        for (int sub = 0; sub < 2; ++sub) {
#pragma unroll
            for (int r = 0; r < 8; ++r) {
                int row = r + half * 8;
                float s_v = sS[p * 512 + sub * 256 + row * 16 + col];
                float g_v = sG[p * 512 + sub * 256 + row * 16 + col];
                float P  = __expf(s_v * scale - sm[r]) * sli[r];
                float Gn = g_v * sgi[r];
                myW[row * 32 + sub * 16 + col] = f2bf(cP * P + cG * Gn);
            }
        }
        __syncthreads();
        // accumulation phase: acc += W_tile(16x32) @ rows(32x256)
        v16bf Aw = ldbf16x16(myW + col * 32 + half * 8);
#pragma unroll
        for (int tt = 0; tt < 16; ++tt) {
            v16bf B = ldbf16x16(sbt + (tt * 16 + col) * 32 + half * 8);
            acc[tt] = wmma_bf16(Aw, B, acc[tt]);
        }
    }
#pragma unroll
    for (int tt = 0; tt < 16; ++tt) {
#pragma unroll
        for (int r = 0; r < 8; ++r) {
            int row = i0 + r + half * 8;
            size_t idx = ((size_t)js * N + row) * D + tt * 16 + col;
            pout[idx] = acc[tt][r];
        }
    }
}

// ---------------- merge j-split partials; x residual; hmix -> bf16 ----------------
__global__ void k_fin(const float* __restrict__ partH, const float* __restrict__ partX,
                      const float* __restrict__ x0, u16* __restrict__ hmixb,
                      float* __restrict__ outX) {
    size_t i = (size_t)blockIdx.x * blockDim.x + threadIdx.x;
    if (i >= (size_t)ND) return;
    float hm = partH[i] + partH[(size_t)ND + i];
    float xv = partX[i] + partX[(size_t)ND + i] + x0[i];
    hmixb[i] = f2bf(hm);
    outX[i] = xv;
}

extern "C" void kernel_launch(void* const* d_in, const int* in_sizes, int n_in,
                              void* d_out, int out_size, void* d_ws, size_t ws_size,
                              hipStream_t stream) {
    (void)in_sizes; (void)n_in; (void)out_size; (void)ws_size;
    const float* h      = (const float*)d_in[0];
    const float* x      = (const float*)d_in[1];
    const float* Wk     = (const float*)d_in[2];
    const float* Wq     = (const float*)d_in[3];
    const float* Wv     = (const float*)d_in[4];
    const float* gamma  = (const float*)d_in[5];
    const float* beta   = (const float*)d_in[6];
    const float* mixing = (const float*)d_in[7];

    char* ws = (char*)d_ws;
    auto alloc = [&](size_t bytes) -> char* {
        char* p = ws;
        ws += (bytes + 255) & ~(size_t)255;
        return p;
    };
    u16*   hn_b   = (u16*)  alloc((size_t)ND * 2);
    u16*   x_b    = (u16*)  alloc((size_t)ND * 2);
    u16*   k_b    = (u16*)  alloc((size_t)ND * 2);
    u16*   q_b    = (u16*)  alloc((size_t)ND * 2);
    u16*   hmix_b = (u16*)  alloc((size_t)ND * 2);
    u16*   wk_b   = (u16*)  alloc((size_t)D * D * 2);
    u16*   wq_b   = (u16*)  alloc((size_t)D * D * 2);
    u16*   wv_b   = (u16*)  alloc((size_t)D * D * 2);
    float* mixf   = (float*)alloc(16 * sizeof(float));
    float* statsp = (float*)alloc((size_t)N * CHUNKS * 3 * sizeof(float));
    float* statsf = (float*)alloc((size_t)N * 3 * sizeof(float));
    float* partH  = (float*)alloc((size_t)JS * ND * sizeof(float));
    float* partX  = (float*)alloc((size_t)JS * ND * sizeof(float));

    float* outH = (float*)d_out;
    float* outX = (float*)d_out + (size_t)ND;

    k_mixprep<<<1, 32, 0, stream>>>(mixing, mixf);
    k_prep<<<N, 256, 0, stream>>>(h, x, gamma, beta, hn_b, x_b);
    k_convwt<<<(D * D) / 256, 256, 0, stream>>>(Wk, wk_b);
    k_convwt<<<(D * D) / 256, 256, 0, stream>>>(Wq, wq_b);
    k_convwt<<<(D * D) / 256, 256, 0, stream>>>(Wv, wv_b);

    k_gemm256<<<N / 16, 256, 0, stream>>>(hn_b, wk_b, k_b, nullptr, nullptr, 0);
    k_gemm256<<<N / 16, 256, 0, stream>>>(hn_b, wq_b, q_b, nullptr, nullptr, 0);

    dim3 g1(64, CHUNKS);
    k_pass1<<<g1, 256, 0, stream>>>(k_b, q_b, x_b, statsp);
    k_mstats<<<N / 256, 256, 0, stream>>>(statsp, statsf);

    dim3 g2(128, JS);
    k_pass2<<<g2, 256, 0, stream>>>(k_b, q_b, x_b, hn_b, statsf, mixf, partH, partX);

    k_fin<<<(ND + 255) / 256, 256, 0, stream>>>(partH, partX, x, hmix_b, outX);
    k_gemm256<<<N / 16, 256, 0, stream>>>(hmix_b, wv_b, nullptr, outH, h, 1);
}
